// RelationModule_12498354831764
// MI455X (gfx1250) — matrix operand order repaired and checked by reading, not compiled, not run on hardware
//
#include <hip/hip_runtime.h>
#include <hip/hip_bf16.h>
#include <math.h>

typedef _Float16 half_t;
typedef _Float16 v16h __attribute__((ext_vector_type(16)));
typedef _Float16 v8h  __attribute__((ext_vector_type(8)));
typedef float    v8f  __attribute__((ext_vector_type(8)));
typedef unsigned int u32x4 __attribute__((ext_vector_type(4)));
typedef int          i32x4 __attribute__((ext_vector_type(4)));
typedef int          i32x8 __attribute__((ext_vector_type(8)));

#define BSZ   8
#define ROI   512
#define DDIM  1024
#define NGRP  16
#define DH    64
#define NGEO  64
#define NTOT  (BSZ*ROI)

static __device__ __forceinline__ float redmax16(float v) {
#pragma unroll
  for (int m = 1; m < 16; m <<= 1) v = fmaxf(v, __shfl_xor(v, m, 32));
  return v;
}
static __device__ __forceinline__ float redsum16(float v) {
#pragma unroll
  for (int m = 1; m < 16; m <<= 1) v += __shfl_xor(v, m, 32);
  return v;
}
static __device__ __forceinline__ float fast_tanh(float s) {
  float e = __expf(2.f * fabsf(s));
  float t = 1.f - 2.f / (e + 1.f);
  return copysignf(t, s);
}

// TDM: 2D tile load global->LDS, data_size=8B. Descriptor per ISA 08 §8.3/8.4.
static __device__ __forceinline__ void tdm_load_2d(
    unsigned int lds_off, const void* gptr, int tile_units, int tile_rows,
    int row_stride_units) {
  unsigned long long ga = (unsigned long long)(uintptr_t)gptr;
  u32x4 g0;
  g0[0] = 1u;                                   // count=1, user mode
  g0[1] = lds_off;                              // lds_addr (bytes)
  g0[2] = (unsigned)(ga & 0xFFFFFFFFu);         // global_addr[31:0]
  g0[3] = (unsigned)((ga >> 32) & 0x01FFFFFFu)  // global_addr[56:32]
          | 0x80000000u;                        // type=2 ("image")
  i32x8 g1;
  g1[0] = (3 << 16);                            // data_size = 8B
  g1[1] = (tile_units & 0xFFFF) << 16;          // tensor_dim0[15:0]
  g1[2] = (tile_rows & 0xFFFF) << 16;           // dim0[31:16]=0 | tensor_dim1[15:0]
  g1[3] = (tile_units & 0xFFFF) << 16;          // dim1[31:16]=0 | tile_dim0
  g1[4] = (tile_rows & 0xFFFF);                 // tile_dim1 | tile_dim2=0
  g1[5] = row_stride_units;                     // tensor_dim0_stride[31:0]
  g1[6] = 0;                                    // stride0[47:32] | stride1[15:0]
  g1[7] = 0;                                    // stride1[47:16]
  i32x4 gz = {0, 0, 0, 0};
#if defined(__clang_major__) && (__clang_major__ >= 23)
  i32x8 gz8 = {0, 0, 0, 0, 0, 0, 0, 0};
  __builtin_amdgcn_tensor_load_to_lds(g0, g1, gz, gz, gz8, 0);
#else
  __builtin_amdgcn_tensor_load_to_lds(g0, g1, gz, gz, 0);
#endif
}

// ---------------------------------------------------------------------------
// Kernel A: pairwise geometric encoding -> tanh(enc@Wgeo^T) -> relu(conv 64->16)
//           -> log(max(.,1e-6)).  Output geoLog[b][g][i][j].
// ---------------------------------------------------------------------------
__global__ __launch_bounds__(256) void rm_geo_kernel(
    const float* __restrict__ boxes, const float* __restrict__ Wgeo,
    const float* __restrict__ bgeo, const float* __restrict__ Wbgc,
    const float* __restrict__ bbgc, float* __restrict__ geoLog) {
  __shared__ float sWg[NGEO * 4];
  __shared__ float sbg[NGEO];
  __shared__ float sWb[NGRP * NGEO];
  __shared__ float sbb[NGRP];
  int t = threadIdx.x;
  for (int i = t; i < NGEO * 4; i += 256) sWg[i] = Wgeo[i];
  for (int i = t; i < NGEO; i += 256) sbg[i] = bgeo[i];
  for (int i = t; i < NGRP * NGEO; i += 256) sWb[i] = Wbgc[i];
  for (int i = t; i < NGRP; i += 256) sbb[i] = bbgc[i];
  __syncthreads();

  long tid = (long)blockIdx.x * 256 + t;
  int j = (int)(tid & 511);
  int i = (int)((tid >> 9) & 511);
  int b = (int)(tid >> 18);

  const float* bi = boxes + (long)(b * ROI + i) * 4;
  const float* bj = boxes + (long)(b * ROI + j) * 4;
  float x1i = bi[0], y1i = bi[1], x2i = bi[2], y2i = bi[3];
  float x1j = bj[0], y1j = bj[1], x2j = bj[2], y2j = bj[3];
  float wi = x2i - x1i, hgt_i = y2i - y1i;
  float wj = x2j - x1j, hgt_j = y2j - y1j;
  float cxi = 0.5f * (x1i + x2i), cyi = 0.5f * (y1i + y2i);
  float cxj = 0.5f * (x1j + x2j), cyj = 0.5f * (y1j + y2j);
  float invwi = 1.f / wi;
  float dx = __logf(fmaxf(fabsf((cxi - cxj) * invwi), 1e-3f));
  float dy = __logf(fmaxf(fabsf((cyi - cyj) * invwi), 1e-3f));
  float dw = __logf(wi / wj);
  float dhh = __logf(hgt_i / hgt_j);

  float acc[NGRP];
#pragma unroll
  for (int g = 0; g < NGRP; g++) acc[g] = sbb[g];
  for (int k = 0; k < NGEO; k++) {
    float s = sWg[k * 4 + 0] * dx + sWg[k * 4 + 1] * dy +
              sWg[k * 4 + 2] * dw + sWg[k * 4 + 3] * dhh + sbg[k];
    float gk = fast_tanh(s);
#pragma unroll
    for (int g = 0; g < NGRP; g++) acc[g] += sWb[g * NGEO + k] * gk;
  }
#pragma unroll
  for (int g = 0; g < NGRP; g++) {
    float v = fmaxf(acc[g], 0.f);
    v = __logf(fmaxf(v, 1e-6f));
    geoLog[((long)(b * NGRP + g) * ROI + i) * ROI + j] = v;
  }
}

// ---------------------------------------------------------------------------
// Kernel B: GEMM  out[row,col] = sum_d X[row,d] * W[col*1024+d] + bias[col]
// stored as f16 remapped to [b][g][n][dh]  (col = g*64+dh, row = b*512+n).
// Block tile 128x64, 8 waves, each wave 16 rows x 64 cols, WMMA f16->f32.
// ---------------------------------------------------------------------------
__global__ __launch_bounds__(256) void rm_proj_kernel(
    const float* __restrict__ X, const float* __restrict__ W,
    const float* __restrict__ bias, half_t* __restrict__ Out) {
  __shared__ __align__(32) half_t As[128 * 40];   // [row][k] stride 40
  __shared__ __align__(32) half_t Bst[64 * 32];   // [col][k] (transposed)
  int tid = threadIdx.x;
  int wave = tid >> 5, lane = tid & 31;
  int hi = lane >> 4, lrow = lane & 15;
  int colBlock = blockIdx.x * 64;
  int rowBlock = blockIdx.y * 128;

  v8f c[4] = {};
  for (int k0 = 0; k0 < DDIM; k0 += 32) {
    {  // A tile: 128 rows x 32 k, f32 -> f16
      int r = tid >> 1;
      int kk = (tid & 1) * 16;
      const float* src = X + (long)(rowBlock + r) * DDIM + k0 + kk;
      half_t* dst = &As[r * 40 + kk];
#pragma unroll
      for (int e = 0; e < 16; e++) dst[e] = (half_t)src[e];
    }
    {  // B tile transposed: Bst[col][d] = W[(colBlock+col)*1024 + k0 + d]
      int col = tid >> 2;
      int d0 = (tid & 3) * 8;
      const float* src = W + (long)(colBlock + col) * DDIM + k0 + d0;
      half_t* dst = &Bst[col * 32 + d0];
#pragma unroll
      for (int e = 0; e < 8; e++) dst[e] = (half_t)src[e];
    }
    __syncthreads();

    v16h a;
    {  // A fragment, contiguous k = 16*hi + e
      const half_t* base = &As[(wave * 16 + lrow) * 40];
      v8h c0 = *(const v8h*)(base + 16 * hi);
      v8h c1 = *(const v8h*)(base + 16 * hi + 8);
#pragma unroll
      for (int e = 0; e < 8; e++) { a[e] = c0[e]; a[8 + e] = c1[e]; }
    }
#pragma unroll
    for (int t4 = 0; t4 < 4; t4++) {
      v16h bf = *(const v16h*)&Bst[(t4 * 16 + lrow) * 32 + hi * 16];
      c[t4] = __builtin_amdgcn_wmma_f32_16x16x32_f16(false, a, false, bf,
                                                     (short)0, c[t4], false, false);
    }
    __syncthreads();
  }

#pragma unroll
  for (int t4 = 0; t4 < 4; t4++) {
    int col = colBlock + t4 * 16 + lrow;
    float bv = bias[col];
    int g = col >> 6, dh = col & 63;
#pragma unroll
    for (int v = 0; v < 8; v++) {
      int row = rowBlock + wave * 16 + v + 8 * hi;
      int bb = row >> 9, n = row & 511;
      Out[((long)(bb * NGRP + g) * ROI + n) * DH + dh] = (half_t)(c[t4][v] + bv);
    }
  }
}

// ---------------------------------------------------------------------------
// Kernel C: fused attention per (b,g,n-tile).  Online softmax over m=512 with
// log-geo bias; O += exp(S-m) @ Y where Y = x@Wgc[g]^T + b_gc (bias folds in
// since softmax rows sum to 1).  Finalize: relu(O / l).
// K chunk arrives via the Tensor Data Mover (wave 0 issues, TENSORcnt wait).
// ---------------------------------------------------------------------------
__global__ __launch_bounds__(256) void rm_attn_kernel(
    const half_t* __restrict__ Qh, const half_t* __restrict__ Kh,
    const half_t* __restrict__ Yh, const float* __restrict__ geoLog,
    float* __restrict__ Out) {
  __shared__ __align__(32) half_t Kls[128 * 64];   // [m][d]
  __shared__ __align__(32) half_t Ylst[64 * 128];  // [o][m] transposed
  __shared__ __align__(32) half_t Pls[8][16 * 32]; // per-wave P scratch

  int tid = threadIdx.x, wave = tid >> 5, lane = tid & 31;
  int hi = lane >> 4, lrow = lane & 15;
  int ntile = blockIdx.x, g = blockIdx.y, b = blockIdx.z;
  long bg = (long)(b * NGRP + g);
  const half_t* Qg = Qh + bg * ROI * DH;
  const half_t* Kg = Kh + bg * ROI * DH;
  const half_t* Yg = Yh + bg * ROI * DH;
  const float* Gl = geoLog + bg * ROI * ROI;
  int nBase = ntile * 128 + wave * 16;
  unsigned int ldsK = (unsigned int)(uintptr_t)(&Kls[0]);  // addr[31:0] == LDS offset

  v16h a0, a1;
  {  // Q fragments (K = 0..31 and 32..63)
    const half_t* qr = Qg + (long)(nBase + lrow) * DH;
    v8h q0 = *(const v8h*)(qr + 16 * hi);
    v8h q1 = *(const v8h*)(qr + 16 * hi + 8);
    v8h q2 = *(const v8h*)(qr + 32 + 16 * hi);
    v8h q3 = *(const v8h*)(qr + 40 + 16 * hi);
#pragma unroll
    for (int e = 0; e < 8; e++) {
      a0[e] = q0[e]; a0[8 + e] = q1[e];
      a1[e] = q2[e]; a1[8 + e] = q3[e];
    }
  }

  float mrun[8], lrun[8];
  v8f c[4] = {};
#pragma unroll
  for (int v = 0; v < 8; v++) { mrun[v] = -3.0e38f; lrun[v] = 0.f; }

  for (int mc = 0; mc < ROI; mc += 128) {
    __syncthreads();
    if (wave == 0) {
      // K chunk via Tensor Data Mover: 128 rows x 128B (16 x 8B units)
      tdm_load_2d(ldsK, Kg + (long)mc * DH, 16, 128, 16);
    }
    {  // Y chunk transposed: Ylst[o][m] (manual; TDM cannot transpose 16-bit)
      for (int idx = tid; idx < 8192; idx += 256) {
        int m = idx >> 6, d = idx & 63;
        Ylst[d * 128 + m] = Yg[(long)(mc + m) * DH + d];
      }
    }
    __builtin_prefetch(&Gl[(long)(nBase + lrow) * ROI + mc], 0, 0);
    if (wave == 0) __builtin_amdgcn_s_wait_tensorcnt(0);
    __syncthreads();

#pragma unroll
    for (int ms = 0; ms < 128; ms += 32) {
      int m0 = mc + ms;
      // logits tiles S0 (m cols 0..15), S1 (16..31): S = Q(16x64) @ K^T
      v16h b0a = *(const v16h*)&Kls[(ms + lrow) * 64 + hi * 16];
      v16h b0b = *(const v16h*)&Kls[(ms + lrow) * 64 + 32 + hi * 16];
      v16h b1a = *(const v16h*)&Kls[(ms + 16 + lrow) * 64 + hi * 16];
      v16h b1b = *(const v16h*)&Kls[(ms + 16 + lrow) * 64 + 32 + hi * 16];
      v8f s0 = {}, s1 = {};
      s0 = __builtin_amdgcn_wmma_f32_16x16x32_f16(false, a0, false, b0a, (short)0, s0, false, false);
      s0 = __builtin_amdgcn_wmma_f32_16x16x32_f16(false, a1, false, b0b, (short)0, s0, false, false);
      s1 = __builtin_amdgcn_wmma_f32_16x16x32_f16(false, a0, false, b1a, (short)0, s1, false, false);
      s1 = __builtin_amdgcn_wmma_f32_16x16x32_f16(false, a1, false, b1b, (short)0, s1, false, false);

      float w0[8], w1[8];
#pragma unroll
      for (int v = 0; v < 8; v++) {
        int n = nBase + v + 8 * hi;
        w0[v] = s0[v] * 0.125f + Gl[(long)n * ROI + m0 + lrow];
        w1[v] = s1[v] * 0.125f + Gl[(long)n * ROI + m0 + 16 + lrow];
      }
      float alpha[8];
#pragma unroll
      for (int v = 0; v < 8; v++) {
        float rm = redmax16(fmaxf(w0[v], w1[v]));
        float mnew = fmaxf(mrun[v], rm);
        alpha[v] = __expf(mrun[v] - mnew);
        mrun[v] = mnew;
      }
      float p0[8], p1[8];
#pragma unroll
      for (int v = 0; v < 8; v++) {
        p0[v] = __expf(w0[v] - mrun[v]);
        p1[v] = __expf(w1[v] - mrun[v]);
        float rs = redsum16(p0[v] + p1[v]);
        lrun[v] = alpha[v] * lrun[v] + rs;
      }
#pragma unroll
      for (int t4 = 0; t4 < 4; t4++)
#pragma unroll
        for (int v = 0; v < 8; v++) c[t4][v] *= alpha[v];

      // C-layout -> A-layout conversion of P via per-wave LDS bounce
      half_t* pw = &Pls[wave][0];
#pragma unroll
      for (int v = 0; v < 8; v++) {
        pw[(v + 8 * hi) * 32 + lrow] = (half_t)p0[v];
        pw[(v + 8 * hi) * 32 + 16 + lrow] = (half_t)p1[v];
      }
      asm volatile("s_wait_dscnt 0" ::: "memory");
      v16h pa;
      {
        v8h c0v = *(const v8h*)&pw[lrow * 32 + 16 * hi];
        v8h c1v = *(const v8h*)&pw[lrow * 32 + 16 * hi + 8];
#pragma unroll
        for (int e = 0; e < 8; e++) { pa[e] = c0v[e]; pa[8 + e] = c1v[e]; }
      }
      // O(16x64) += P(16x32) @ Y(32x64)
#pragma unroll
      for (int t4 = 0; t4 < 4; t4++) {
        v16h yf = *(const v16h*)&Ylst[(t4 * 16 + lrow) * 128 + ms + hi * 16];
        c[t4] = __builtin_amdgcn_wmma_f32_16x16x32_f16(false, pa, false, yf,
                                                       (short)0, c[t4], false, false);
      }
    }
  }

#pragma unroll
  for (int t4 = 0; t4 < 4; t4++) {
#pragma unroll
    for (int v = 0; v < 8; v++) {
      int n = nBase + v + 8 * hi;
      float val = fmaxf(c[t4][v] / lrun[v], 0.f);
      Out[(long)(b * ROI + n) * DDIM + g * DH + t4 * 16 + lrow] = val;
    }
  }
}

// ---------------------------------------------------------------------------
extern "C" void kernel_launch(void* const* d_in, const int* in_sizes, int n_in,
                              void* d_out, int out_size, void* d_ws, size_t ws_size,
                              hipStream_t stream) {
  const float* x     = (const float*)d_in[0];
  const float* boxes = (const float*)d_in[1];
  const float* Wgeo  = (const float*)d_in[2];
  const float* bgeo  = (const float*)d_in[3];
  const float* Wbgc  = (const float*)d_in[4];
  const float* bbgc  = (const float*)d_in[5];
  const float* Wq    = (const float*)d_in[6];
  const float* bq    = (const float*)d_in[7];
  const float* Wk    = (const float*)d_in[8];
  const float* bk    = (const float*)d_in[9];
  const float* Wgc   = (const float*)d_in[10];
  const float* bgc   = (const float*)d_in[11];
  float* out = (float*)d_out;

  char* ws = (char*)d_ws;
  size_t geoBytes = (size_t)BSZ * NGRP * ROI * ROI * sizeof(float);  // 134 MB
  size_t projBytes = (size_t)BSZ * NGRP * ROI * DH * sizeof(half_t); // 8.4 MB
  float* geoLog = (float*)ws;
  half_t* QhP = (half_t*)(ws + geoBytes);
  half_t* KhP = (half_t*)(ws + geoBytes + projBytes);
  half_t* YhP = (half_t*)(ws + geoBytes + 2 * projBytes);

  rm_geo_kernel<<<(BSZ * ROI * ROI) / 256, 256, 0, stream>>>(
      boxes, Wgeo, bgeo, Wbgc, bbgc, geoLog);

  dim3 gProj(DDIM / 64, NTOT / 128);
  rm_proj_kernel<<<gProj, 256, 0, stream>>>(x, Wq, bq, QhP);
  rm_proj_kernel<<<gProj, 256, 0, stream>>>(x, Wk, bk, KhP);
  rm_proj_kernel<<<gProj, 256, 0, stream>>>(x, Wgc, bgc, YhP);  // Y folds b_gc

  dim3 gAttn(ROI / 128, NGRP, BSZ);
  rm_attn_kernel<<<gAttn, 256, 0, stream>>>(QhP, KhP, YhP, geoLog, out);
}
